// SelectorBlock_77309411328334
// MI455X (gfx1250) — matrix operand order repaired and checked by reading, not compiled, not run on hardware
//
#include <hip/hip_runtime.h>
#include <hip/hip_bf16.h>

// ---------------------------------------------------------------------------
// CDNA5 (gfx1250) implementation of the cost-aware hetero-MoE transformer
// block. Dense linear algebra -> v_wmma_f32_16x16x32_f16 (fp32 accumulate),
// K-tile staging -> global_load_async_to_lds_b128 double-buffered in LDS.
// ---------------------------------------------------------------------------

typedef __attribute__((ext_vector_type(16))) _Float16 v16h;
typedef __attribute__((ext_vector_type(8)))  _Float16 v8h;
typedef __attribute__((ext_vector_type(4)))  _Float16 v4h;
typedef __attribute__((ext_vector_type(8)))  float    v8f;

__device__ __forceinline__ float geluf(float x) {
  return 0.5f * x * (1.0f + erff(x * 0.7071067811865476f));
}

// LDS byte offset of a __shared__ address: low 32 bits of the flat address
// (LDS aperture maps addr[31:0] -> LDS offset, ISA 10.2).
__device__ __forceinline__ unsigned lds_off(const void* p) {
  return (unsigned)(unsigned long long)p;
}

// Async copy of 16B global -> LDS, tracked by ASYNCcnt (ISA 10.7 / 15.18.3).
__device__ __forceinline__ void async_copy_b128(unsigned lds_addr, const void* gptr) {
  asm volatile("global_load_async_to_lds_b128 %0, %1, off"
               :: "v"(lds_addr), "v"((unsigned long long)gptr) : "memory");
}
__device__ __forceinline__ void wait_asynccnt0() {
  asm volatile("s_wait_asynccnt 0x0" ::: "memory");
}

// ---------------------------------------------------------------------------
// WMMA GEMM: C[M,N] = A[M,K] * op(B) + bias[N]   (bias always valid)
//   B_IS_KN == false : B stored [N,K] row-major (torch weight layout, A @ B^T)
//   B_IS_KN == true  : B stored [K,N] row-major (A @ B, used for P @ V)
// Block tile 64(M) x 128(N) x 32(K), double-buffered LDS; 8 wave32 waves each
// own a 32x32 output tile -> 4 v_wmma_f32_16x16x32_f16 per wave per K-step.
// Requires: M % 64 == 0, N % 128 == 0, K % 32 == 0 (true for every call here).
// ---------------------------------------------------------------------------
template <bool B_IS_KN>
__global__ __launch_bounds__(256)
void wmma_gemm_kernel(const _Float16* __restrict__ A,
                      const _Float16* __restrict__ Bm,
                      const float* __restrict__ bias,
                      float* __restrict__ C,
                      int M, int N, int K,
                      long long sA, long long sB, long long sC)
{
  constexpr int PA = 40;    // As pitch in halves: 32 + 8 pad (80B rows, 16B aligned)
  constexpr int PB = 136;   // Bs pitch in halves: 128 + 8 pad (272B rows, 16B aligned)
  __shared__ __align__(16) _Float16 As[2][64 * PA]; // [m][k], K-contiguous
  __shared__ __align__(16) _Float16 Bs[2][32 * PB]; // [k][n], N-contiguous

  const int bz = blockIdx.z;
  A  += (long long)bz * sA;
  Bm += (long long)bz * sB;
  C  += (long long)bz * sC;

  const int n0 = blockIdx.x * 128;
  const int m0 = blockIdx.y * 64;

  const int tid  = threadIdx.x;
  const int lane = tid & 31;
  const int wave = tid >> 5;     // 0..7
  const int wm   = wave & 1;     // 32-row slab
  const int wn   = wave >> 1;    // 32-col slab

  // ---- per-thread staging coordinates; pointers bumped by constants ----
  const int ar = tid >> 2;            // 0..63
  const int ac = (tid & 3) * 8;       // 0,8,16,24
  const _Float16* aptr = A + (long long)(m0 + ar) * K + ac;

  int bk, bn;
  const _Float16* bptr;
  if (B_IS_KN) {
    bk = tid >> 3;                    // 0..31
    bn = (tid & 7) * 16;              // 0..112
    bptr = Bm + (long long)bk * N + (n0 + bn);
  } else {
    bn = tid >> 1;                    // 0..127
    bk = (tid & 1) * 16;              // 0,16
    bptr = Bm + (long long)(n0 + bn) * K + bk;
  }
  const long long aStep = 32;
  const long long bStep = B_IS_KN ? (long long)32 * N : 32;

  const unsigned ldsA0 = lds_off(&As[0][ar * PA + ac]);
  const unsigned ldsA1 = lds_off(&As[1][ar * PA + ac]);
  const unsigned ldsB0 = lds_off(&Bs[0][bk * PB + bn]);
  const unsigned ldsB1 = lds_off(&Bs[1][bk * PB + bn]);

  // Stage one K-tile into buffer `buf`, advancing the staging pointers.
  auto stage = [&](int buf) {
    async_copy_b128(buf ? ldsA1 : ldsA0, aptr);          // A: 16B per thread
    if (B_IS_KN) {
      unsigned d = buf ? ldsB1 : ldsB0;                  // B: 2 x 16B per thread
      async_copy_b128(d,      bptr);
      async_copy_b128(d + 16, bptr + 8);
    } else {
      // [N,K] layout: vector-load 32 K-halves of one N row, scatter-transpose.
      v8h b0 = *(const v8h*)bptr;
      v8h b1 = *(const v8h*)(bptr + 8);
      _Float16* bs = &Bs[buf][0];
#pragma unroll
      for (int j = 0; j < 8; ++j) {
        bs[(bk + j)     * PB + bn] = b0[j];
        bs[(bk + j + 8) * PB + bn] = b1[j];
      }
      __builtin_prefetch(bptr + 2 * bStep, 0, 1);        // global_prefetch_b8
    }
    aptr += aStep;
    bptr += bStep;
  };

  v8f acc[2][2] = {};
  const int nIter = K >> 5;

  stage(0);
  wait_asynccnt0();
  __syncthreads();

  for (int it = 0; it < nIter; ++it) {
    const int cur = it & 1;
    if (it + 1 < nIter) stage(cur ^ 1);   // async-fill next buffer during compute

    // A fragment (16x32 MxK): lane&15 -> M row; lanes 0-15 hold K {0..7,16..23},
    // lanes 16-31 hold K {8..15,24..31}. (ISA 7.12.2)
    const int koff = (lane >> 4) ? 8 : 0;
    const int arow = wm * 32 + (lane & 15);
    v16h af[2], bf[2];
#pragma unroll
    for (int mi = 0; mi < 2; ++mi) {
      const _Float16* ap = &As[cur][(arow + mi * 16) * PA];
      v8h lo = *(const v8h*)(ap + koff);
      v8h hi = *(const v8h*)(ap + 16 + koff);
#pragma unroll
      for (int i = 0; i < 8; ++i) { af[mi][i] = lo[i]; af[mi][8 + i] = hi[i]; }
    }
    // B fragment (32x16 KxN): lane&15 -> N column; K order matches A:
    // lanes 0-15 hold K {0..7,16..23}, lanes 16-31 hold K {8..15,24..31}.
    const int bcol = lane & 15;
#pragma unroll
    for (int ni = 0; ni < 2; ++ni) {
      const _Float16* bp = &Bs[cur][wn * 32 + ni * 16 + bcol];
#pragma unroll
      for (int i = 0; i < 8; ++i) { bf[ni][i] = bp[(koff + i) * PB]; bf[ni][8 + i] = bp[(16 + koff + i) * PB]; }
    }
#pragma unroll
    for (int mi = 0; mi < 2; ++mi)
#pragma unroll
      for (int ni = 0; ni < 2; ++ni)
        acc[mi][ni] = __builtin_amdgcn_wmma_f32_16x16x32_f16(
            false, af[mi], false, bf[ni], (short)0, acc[mi][ni], false, false);

    wait_asynccnt0();     // next buffer fully resident in LDS
    __syncthreads();      // one barrier per K-step (double buffered)
  }

  // C/D layout: lane&15 -> N; lanes 0-15 rows M=0..7, lanes 16-31 rows M=8..15.
#pragma unroll
  for (int mi = 0; mi < 2; ++mi) {
    const int mb = m0 + wm * 32 + mi * 16 + ((lane >> 4) ? 8 : 0);
#pragma unroll
    for (int ni = 0; ni < 2; ++ni) {
      const int n = n0 + wn * 32 + ni * 16 + (lane & 15);
      const float bv = bias[n];
      float* cp = C + (long long)mb * N + n;
#pragma unroll
      for (int i = 0; i < 8; ++i)
        cp[(long long)i * N] = acc[mi][ni][i] + bv;
    }
  }
}

// ---------------------------------------------------------------------------
// Elementwise / small kernels
// ---------------------------------------------------------------------------

// RMSNorm per token; writes h (f16) and gelu(h) (f16).
__global__ __launch_bounds__(256)
void rmsnorm_kernel(const float* __restrict__ x, const float* __restrict__ w,
                    _Float16* __restrict__ h16, _Float16* __restrict__ gh16,
                    int D, float inv_sqrt_d)
{
  const long long t = blockIdx.x;
  const float* xp = x + t * D;
  __shared__ float red[256];
  float s = 0.f;
  for (int d = threadIdx.x; d < D; d += 256) { float v = xp[d]; s += v * v; }
  red[threadIdx.x] = s;
  __syncthreads();
  for (int o = 128; o > 0; o >>= 1) {
    if (threadIdx.x < o) red[threadIdx.x] += red[threadIdx.x + o];
    __syncthreads();
  }
  const float inv = 1.0f / (sqrtf(red[0]) * inv_sqrt_d + 1e-8f);
  for (int d = threadIdx.x; d < D; d += 256) {
    float hv = w[d] * xp[d] * inv;
    h16[t * D + d]  = (_Float16)hv;
    gh16[t * D + d] = (_Float16)geluf(hv);
  }
}

// Vectorized fp32 -> f16 cast (4 elems/thread), optional exact-erf GELU.
__global__ __launch_bounds__(256)
void cast_f16v4_kernel(const float4* __restrict__ src, v4h* __restrict__ dst,
                       long long n4, int apply_gelu)
{
  long long i = (long long)blockIdx.x * 256 + threadIdx.x;
  if (i >= n4) return;
  float4 v = src[i];
  if (apply_gelu) { v.x = geluf(v.x); v.y = geluf(v.y); v.z = geluf(v.z); v.w = geluf(v.w); }
  v4h o;
  o[0] = (_Float16)v.x; o[1] = (_Float16)v.y; o[2] = (_Float16)v.z; o[3] = (_Float16)v.w;
  dst[i] = o;
}

// Strided fp32 -> f16, 4-wide (extract Q/K/V slices from the fused qkv buffer).
// cols/4 == (1 << c4shift).
__global__ __launch_bounds__(256)
void cast_strided_f16v4_kernel(const float* __restrict__ src, v4h* __restrict__ dst,
                               long long n4, int src_ld, int src_off, int c4shift)
{
  long long i = (long long)blockIdx.x * 256 + threadIdx.x;
  if (i >= n4) return;
  long long r  = i >> c4shift;
  int       c4 = (int)(i & ((1 << c4shift) - 1)) * 4;
  float4 v = *(const float4*)(src + r * src_ld + src_off + c4);
  v4h o;
  o[0] = (_Float16)v.x; o[1] = (_Float16)v.y; o[2] = (_Float16)v.z; o[3] = (_Float16)v.w;
  dst[i] = o;
}

// Row softmax with scale, fp32 scores -> f16 probabilities. One block per row.
__global__ __launch_bounds__(256)
void softmax_rows_kernel(const float* __restrict__ S, _Float16* __restrict__ P16,
                         int T, float scale)
{
  const long long r = blockIdx.x;
  const float* sp = S + r * T;
  _Float16* pp = P16 + r * T;
  __shared__ float red[256];

  float m = -3.4e38f;
  for (int i = threadIdx.x; i < T; i += 256) m = fmaxf(m, sp[i] * scale);
  red[threadIdx.x] = m;
  __syncthreads();
  for (int o = 128; o > 0; o >>= 1) {
    if (threadIdx.x < o) red[threadIdx.x] = fmaxf(red[threadIdx.x], red[threadIdx.x + o]);
    __syncthreads();
  }
  const float rowmax = red[0];
  __syncthreads();

  float s = 0.f;
  for (int i = threadIdx.x; i < T; i += 256) s += __expf(sp[i] * scale - rowmax);
  red[threadIdx.x] = s;
  __syncthreads();
  for (int o = 128; o > 0; o >>= 1) {
    if (threadIdx.x < o) red[threadIdx.x] += red[threadIdx.x + o];
    __syncthreads();
  }
  const float inv = 1.0f / red[0];
  for (int i = threadIdx.x; i < T; i += 256)
    pp[i] = (_Float16)(__expf(sp[i] * scale - rowmax) * inv);
}

// Router: logits -> softmax -> top-2 -> gate (softmax over the 2 top probs).
__global__ __launch_bounds__(256)
void router_kernel(const _Float16* __restrict__ h16, const float* __restrict__ Wr,
                   const float* __restrict__ br, const float* __restrict__ cost,
                   int* __restrict__ topi, float* __restrict__ gate,
                   long long BT, int D)
{
  long long t = (long long)blockIdx.x * 256 + threadIdx.x;
  if (t >= BT) return;
  float lg[8];
  const _Float16* hp = h16 + t * D;
#pragma unroll
  for (int e = 0; e < 8; ++e) {
    const float* wp = Wr + (long long)e * D;
    float s = 0.f;
    for (int d = 0; d < D; ++d) s += (float)hp[d] * wp[d];
    lg[e] = s + br[e] - 5e-4f * cost[e];
  }
  float m = lg[0];
#pragma unroll
  for (int e = 1; e < 8; ++e) m = fmaxf(m, lg[e]);
  float Z = 0.f, p[8];
#pragma unroll
  for (int e = 0; e < 8; ++e) { p[e] = __expf(lg[e] - m); Z += p[e]; }
#pragma unroll
  for (int e = 0; e < 8; ++e) p[e] /= Z;
  int i0 = 0;
#pragma unroll
  for (int e = 1; e < 8; ++e) if (p[e] > p[i0]) i0 = e;
  int i1 = (i0 == 0) ? 1 : 0;
#pragma unroll
  for (int e = 0; e < 8; ++e) if (e != i0 && p[e] > p[i1]) i1 = e;
  float e1 = __expf(p[i1] - p[i0]);     // gate = softmax([p_i0, p_i1])
  float g0 = 1.0f / (1.0f + e1);
  topi[t * 2 + 0] = i0;  topi[t * 2 + 1] = i1;
  gate[t * 2 + 0] = g0;  gate[t * 2 + 1] = 1.0f - g0;
}

// c0[j] = expert(0)[j] = gelu(b1) @ W2^T + b2 for one expert.
__global__ __launch_bounds__(256)
void c0_kernel(const float* __restrict__ b1, const float* __restrict__ W2,
               const float* __restrict__ b2, int hid, float* __restrict__ c0)
{
  int j = blockIdx.x * 256 + threadIdx.x;
  if (j >= 512) return;
  float s = b2[j];
  const float* wp = W2 + (long long)j * hid;
  for (int i = 0; i < hid; ++i) s += geluf(b1[i]) * wp[i];
  c0[j] = s;
}

__global__ __launch_bounds__(256)
void csum_kernel(const float* __restrict__ c0, float* __restrict__ csum)
{
  int j = blockIdx.x * 256 + threadIdx.x;
  if (j >= 512) return;
  float s = 0.f;
#pragma unroll
  for (int e = 0; e < 8; ++e) s += c0[e * 512 + j];
  csum[j] = s;
}

// acc += gate * expert_out for tokens routed to expert e.
__global__ __launch_bounds__(256)
void moe_accum_kernel(const float* __restrict__ eo, const int* __restrict__ topi,
                      const float* __restrict__ gate, float* __restrict__ acc,
                      int e, long long n)
{
  long long i = (long long)blockIdx.x * 256 + threadIdx.x;
  if (i >= n) return;
  long long t = i >> 9;         // / 512
  float g = 0.f;
  if (topi[t * 2 + 0] == e) g += gate[t * 2 + 0];
  if (topi[t * 2 + 1] == e) g += gate[t * 2 + 1];
  if (g != 0.f) acc[i] += g * eo[i];
}

// moe16 = acc + sum_k gate_k*(csum - c0[topi_k]) + 0.1*shared   (f16 for Wu GEMM)
__global__ __launch_bounds__(256)
void moe_final_kernel(const float* __restrict__ acc, const int* __restrict__ topi,
                      const float* __restrict__ gate, const float* __restrict__ c0,
                      const float* __restrict__ csum, const float* __restrict__ sh,
                      _Float16* __restrict__ moe16, long long n)
{
  long long i = (long long)blockIdx.x * 256 + threadIdx.x;
  if (i >= n) return;
  long long t = i >> 9;
  int j = (int)(i & 511);
  int i0 = topi[t * 2 + 0], i1 = topi[t * 2 + 1];
  float extra = gate[t * 2 + 0] * (csum[j] - c0[i0 * 512 + j])
              + gate[t * 2 + 1] * (csum[j] - c0[i1 * 512 + j]);
  moe16[i] = (_Float16)(acc[i] + extra + 0.1f * sh[i]);
}

// out = x + attn + up + core (float4 vectorized)
__global__ __launch_bounds__(256)
void final_sum_kernel(const float4* __restrict__ x, const float4* __restrict__ attn,
                      const float4* __restrict__ up, const float4* __restrict__ core,
                      float4* __restrict__ out, long long n4)
{
  long long i = (long long)blockIdx.x * 256 + threadIdx.x;
  if (i >= n4) return;
  float4 a = x[i], b = attn[i], c = up[i], d = core[i];
  float4 o;
  o.x = a.x + b.x + c.x + d.x;
  o.y = a.y + b.y + c.y + d.y;
  o.z = a.z + b.z + c.z + d.z;
  o.w = a.w + b.w + c.w + d.w;
  out[i] = o;
}

// ---------------------------------------------------------------------------
// Host orchestration
// ---------------------------------------------------------------------------
extern "C" void kernel_launch(void* const* d_in, const int* in_sizes, int n_in,
                              void* d_out, int out_size, void* d_ws, size_t ws_size,
                              hipStream_t stream)
{
  (void)in_sizes; (void)n_in; (void)out_size; (void)ws_size;

  const int Bb = 4, T = 2048, D = 1024, LAT = 512, E = 8;
  const long long BT = (long long)Bb * T;      // 8192 tokens
  const int HIDS[8] = {1024, 2048, 4096, 1024, 2048, 4096, 1024, 2048};
  const int MAXHID = 4096;

  // ---- inputs (setup_inputs dict order, expert_params flattened in order) ----
  const float* x      = (const float*)d_in[0];
  const float* w_norm = (const float*)d_in[1];
  const float* Wqkv   = (const float*)d_in[2];
  const float* Wo     = (const float*)d_in[3];
  const float* Wd     = (const float*)d_in[4];
  const float* bd     = (const float*)d_in[5];
  const float* Wu     = (const float*)d_in[6];
  const float* bu     = (const float*)d_in[7];
  const float* Wr     = (const float*)d_in[8];
  const float* br     = (const float*)d_in[9];
  const float *eW1[8], *eb1[8], *eW2[8], *eb2[8];
  for (int e = 0; e < 8; ++e) {
    eW1[e] = (const float*)d_in[10 + e * 4 + 0];
    eb1[e] = (const float*)d_in[10 + e * 4 + 1];
    eW2[e] = (const float*)d_in[10 + e * 4 + 2];
    eb2[e] = (const float*)d_in[10 + e * 4 + 3];
  }
  const float* Ws1  = (const float*)d_in[42];
  const float* bs1  = (const float*)d_in[43];
  const float* Ws2  = (const float*)d_in[44];
  const float* bs2  = (const float*)d_in[45];
  const float* Wc   = (const float*)d_in[46];
  const float* bc   = (const float*)d_in[47];
  const float* cost = (const float*)d_in[48];

  // ---- workspace bump allocator (aliased arenas for the big transients) ----
  char* wp = (char*)d_ws;
  auto alloc = [&](size_t bytes) -> char* {
    char* p = wp; wp += (bytes + 255) & ~(size_t)255; return p;
  };

  // Arena A: qkv fp32 (96MB) reused later for per-expert hidden fp32 (<=128MB).
  char* arenaA = alloc((size_t)BT * MAXHID * sizeof(float));
  float*     qkv32 = (float*)arenaA;
  float*     hid32 = (float*)arenaA;
  // Arena B: attention scores fp32 (67MB) -> expert hidden f16 -> up32+core32.
  char* arenaB = alloc((size_t)Bb * T * T * sizeof(float));
  float*     S32    = (float*)arenaB;
  _Float16*  hid16  = (_Float16*)arenaB;
  float*     up32   = (float*)arenaB;
  float*     core32 = (float*)(arenaB + (size_t)BT * D * sizeof(float));

  _Float16* h16    = (_Float16*)alloc((size_t)BT * D * 2);
  _Float16* gh16   = (_Float16*)alloc((size_t)BT * D * 2);
  _Float16* q16    = (_Float16*)alloc((size_t)BT * D * 2);
  _Float16* k16    = (_Float16*)alloc((size_t)BT * D * 2);
  _Float16* v16    = (_Float16*)alloc((size_t)BT * D * 2);
  _Float16* P16    = (_Float16*)alloc((size_t)Bb * T * T * 2);
  float*    O32    = (float*)alloc((size_t)BT * D * 4);
  _Float16* O16    = (_Float16*)alloc((size_t)BT * D * 2);
  float*    attn32 = (float*)alloc((size_t)BT * D * 4);
  float*    lat32  = (float*)alloc((size_t)BT * LAT * 4);
  _Float16* glat16 = (_Float16*)alloc((size_t)BT * LAT * 2);
  float*    eo32   = (float*)alloc((size_t)BT * LAT * 4);
  float*    acc32  = (float*)alloc((size_t)BT * LAT * 4);
  float*    s1_32  = (float*)alloc((size_t)BT * LAT * 4);
  _Float16* gs1_16 = (_Float16*)alloc((size_t)BT * LAT * 2);
  float*    sh32   = (float*)alloc((size_t)BT * LAT * 4);
  _Float16* moe16  = (_Float16*)alloc((size_t)BT * LAT * 2);
  int*      topi   = (int*)alloc((size_t)BT * 2 * 4);
  float*    gate   = (float*)alloc((size_t)BT * 2 * 4);
  float*    c0     = (float*)alloc((size_t)E * LAT * 4);
  float*    csum   = (float*)alloc((size_t)LAT * 4);
  float*    zbias  = (float*)alloc((size_t)4096 * 4);   // zero bias for bias-less GEMMs

  // f16 weights
  _Float16* Wqkv16 = (_Float16*)alloc((size_t)3 * D * D * 2);
  _Float16* Wo16   = (_Float16*)alloc((size_t)D * D * 2);
  _Float16* Wd16   = (_Float16*)alloc((size_t)LAT * D * 2);
  _Float16* Wu16   = (_Float16*)alloc((size_t)D * LAT * 2);
  _Float16* Ws1_16 = (_Float16*)alloc((size_t)LAT * LAT * 2);
  _Float16* Ws2_16 = (_Float16*)alloc((size_t)LAT * LAT * 2);
  _Float16* Wc16   = (_Float16*)alloc((size_t)D * D * 2);
  _Float16 *eW1_16[8], *eW2_16[8];
  for (int e = 0; e < 8; ++e) {
    eW1_16[e] = (_Float16*)alloc((size_t)HIDS[e] * LAT * 2);
    eW2_16[e] = (_Float16*)alloc((size_t)LAT * HIDS[e] * 2);
  }

  hipMemsetAsync(zbias, 0, 4096 * sizeof(float), stream);

  // ---- launch helpers ----
  auto cast16 = [&](const float* src, _Float16* dst, long long n, int gelu) {
    long long n4 = n >> 2;                        // all sizes are %4 == 0
    unsigned blocks = (unsigned)((n4 + 255) / 256);
    cast_f16v4_kernel<<<dim3(blocks), dim3(256), 0, stream>>>(
        (const float4*)src, (v4h*)dst, n4, gelu);
  };
  auto gemm = [&](const _Float16* A, const _Float16* Bm, const float* bias, float* C,
                  int M, int N, int K, int batch,
                  long long sA, long long sB, long long sC, bool b_kn) {
    dim3 g((unsigned)(N / 128), (unsigned)(M / 64), (unsigned)batch), blk(256);
    const float* bp = bias ? bias : zbias;
    if (b_kn) wmma_gemm_kernel<true ><<<g, blk, 0, stream>>>(A, Bm, bp, C, M, N, K, sA, sB, sC);
    else      wmma_gemm_kernel<false><<<g, blk, 0, stream>>>(A, Bm, bp, C, M, N, K, sA, sB, sC);
  };

  // ---- 0) weight casts to f16 ----
  cast16(Wqkv, Wqkv16, (long long)3 * D * D, 0);
  cast16(Wo,   Wo16,   (long long)D * D,     0);
  cast16(Wd,   Wd16,   (long long)LAT * D,   0);
  cast16(Wu,   Wu16,   (long long)D * LAT,   0);
  cast16(Ws1,  Ws1_16, (long long)LAT * LAT, 0);
  cast16(Ws2,  Ws2_16, (long long)LAT * LAT, 0);
  cast16(Wc,   Wc16,   (long long)D * D,     0);
  for (int e = 0; e < 8; ++e) {
    cast16(eW1[e], eW1_16[e], (long long)HIDS[e] * LAT, 0);
    cast16(eW2[e], eW2_16[e], (long long)LAT * HIDS[e], 0);
  }

  // ---- 1) RMSNorm -> h16, gelu(h) ----
  rmsnorm_kernel<<<dim3((unsigned)BT), dim3(256), 0, stream>>>(
      x, w_norm, h16, gh16, D, 1.0f / 32.0f /* D^-0.5 */);

  // ---- 2) Attention ----
  gemm(h16, Wqkv16, nullptr, qkv32, (int)BT, 3 * D, D, 1, 0, 0, 0, false);
  {
    long long n4 = (BT * D) >> 2;
    unsigned blocks = (unsigned)((n4 + 255) / 256);
    cast_strided_f16v4_kernel<<<dim3(blocks), dim3(256), 0, stream>>>(
        qkv32, (v4h*)q16, n4, 3 * D, 0, 8);
    cast_strided_f16v4_kernel<<<dim3(blocks), dim3(256), 0, stream>>>(
        qkv32, (v4h*)k16, n4, 3 * D, D, 8);
    cast_strided_f16v4_kernel<<<dim3(blocks), dim3(256), 0, stream>>>(
        qkv32, (v4h*)v16, n4, 3 * D, 2 * D, 8);
  }
  // S = Q @ K^T per batch
  gemm(q16, k16, nullptr, S32, T, T, D, Bb,
       (long long)T * D, (long long)T * D, (long long)T * T, false);
  softmax_rows_kernel<<<dim3((unsigned)(Bb * T)), dim3(256), 0, stream>>>(
      S32, P16, T, 1.0f / 32.0f);
  // O = P @ V per batch (B is [K=T, N=D] row-major)
  gemm(P16, v16, nullptr, O32, T, D, T, Bb,
       (long long)T * T, (long long)T * D, (long long)T * D, true);
  cast16(O32, O16, BT * D, 0);
  gemm(O16, Wo16, nullptr, attn32, (int)BT, D, D, 1, 0, 0, 0, false);

  // ---- 3) MoE input / router ----
  gemm(h16, Wd16, bd, lat32, (int)BT, LAT, D, 1, 0, 0, 0, false);
  cast16(lat32, glat16, BT * LAT, 1);   // gelu(lat), shared by all experts
  router_kernel<<<dim3((unsigned)((BT + 255) / 256)), dim3(256), 0, stream>>>(
      h16, Wr, br, cost, topi, gate, BT, D);

  // expert-at-zero corrections
  for (int e = 0; e < 8; ++e)
    c0_kernel<<<dim3(2), dim3(256), 0, stream>>>(eb1[e], eW2[e], eb2[e], HIDS[e], c0 + e * 512);
  csum_kernel<<<dim3(2), dim3(256), 0, stream>>>(c0, csum);

  hipMemsetAsync(acc32, 0, (size_t)BT * LAT * sizeof(float), stream);

  // ---- 4) experts (dense, gated accumulate) ----
  const long long nLat = BT * LAT;
  const unsigned latBlocks = (unsigned)((nLat + 255) / 256);
  for (int e = 0; e < 8; ++e) {
    const int hid = HIDS[e];
    gemm(glat16, eW1_16[e], eb1[e], hid32, (int)BT, hid, LAT, 1, 0, 0, 0, false);
    cast16(hid32, hid16, BT * (long long)hid, 1);  // gelu
    gemm(hid16, eW2_16[e], eb2[e], eo32, (int)BT, LAT, hid, 1, 0, 0, 0, false);
    moe_accum_kernel<<<dim3(latBlocks), dim3(256), 0, stream>>>(eo32, topi, gate, acc32, e, nLat);
  }

  // ---- 5) shared expert ----
  gemm(glat16, Ws1_16, bs1, s1_32, (int)BT, LAT, LAT, 1, 0, 0, 0, false);
  cast16(s1_32, gs1_16, nLat, 1);
  gemm(gs1_16, Ws2_16, bs2, sh32, (int)BT, LAT, LAT, 1, 0, 0, 0, false);

  // ---- 6) combine, up-project, core branch, residual ----
  moe_final_kernel<<<dim3(latBlocks), dim3(256), 0, stream>>>(
      acc32, topi, gate, c0, csum, sh32, moe16, nLat);
  gemm(moe16, Wu16, bu, up32,   (int)BT, D, LAT, 1, 0, 0, 0, false);
  gemm(gh16,  Wc16, bc, core32, (int)BT, D, D,   1, 0, 0, 0, false);

  const long long nOut = BT * D;
  final_sum_kernel<<<dim3((unsigned)((nOut >> 2) + 255) / 256), dim3(256), 0, stream>>>(
      (const float4*)x, (const float4*)attn32, (const float4*)up32,
      (const float4*)core32, (float4*)d_out, nOut >> 2);
}
